// TransFusionHeadAnchorMatching_38689065402458
// MI455X (gfx1250) — matrix operand order repaired and checked by reading, not compiled, not run on hardware
//
#include <hip/hip_runtime.h>
#include <hip/hip_bf16.h>

// ---------------------------------------------------------------------------
// TransFusion head on MI455X (gfx1250).
//  * three 3x3 convs = implicit GEMM on v_wmma_f32_16x16x32_f16 (f32 accum)
//  * f16 working set (~100MB) is L2-resident (192MB): im2col re-reads hit L2
//  * register-staged prefetch of the next K-chunk + global_prefetch for kc+2
//  * wave tiling 4(M) x 2(N): 2 A-frags x 4 B-frags per wave -> 8 WMMAs per
//    24 DS ops (B fragment reused by two consecutive WMMAs)
// ---------------------------------------------------------------------------

typedef _Float16 f16_t;
typedef __attribute__((ext_vector_type(16))) _Float16 v16h;
typedef __attribute__((ext_vector_type(8)))  float    v8f;

#define HB     2          // batch
#define CIN1   512
#define HID    128
#define HH     180
#define WW     180
#define HW     (HH*WW)    // 32400
#define TEXTC  60
#define NPROP  200

__constant__ float c_anchors[10][3] = {
    {4.63f,1.97f,1.74f},{6.93f,2.51f,2.84f},{6.37f,2.85f,3.19f},
    {10.5f,2.94f,3.47f},{12.29f,2.9f,3.87f},{0.5f,2.53f,0.98f},
    {2.11f,0.77f,1.47f},{1.7f,0.6f,1.28f},{0.73f,0.67f,1.77f},
    {0.41f,0.41f,1.07f}};

__device__ __forceinline__ float dsigmoid(float x) { return 1.f / (1.f + __expf(-x)); }

// ---------------------------------------------------------------------------
// f32 -> f16 elementwise convert (grid-stride).
// ---------------------------------------------------------------------------
__global__ void cvt_f16_kernel(const float* __restrict__ src,
                               f16_t* __restrict__ dst, long n)
{
    long i = (long)blockIdx.x * blockDim.x + threadIdx.x;
    long stride = (long)gridDim.x * blockDim.x;
    for (; i < n; i += stride) dst[i] = (f16_t)src[i];
}

// Weights: [COUT x KTOT] f32 -> [COUT_PAD x KTOT] f16, zero-padded rows.
__global__ void cvt_wgt_kernel(const float* __restrict__ src,
                               f16_t* __restrict__ dst,
                               int cout, int coutpad, int ktot)
{
    long n = (long)coutpad * ktot;
    long i = (long)blockIdx.x * blockDim.x + threadIdx.x;
    long stride = (long)gridDim.x * blockDim.x;
    for (; i < n; i += stride) {
        int r = (int)(i / ktot);
        dst[i] = (r < cout) ? (f16_t)src[i] : (f16_t)0.f;
    }
}

// ---------------------------------------------------------------------------
// Implicit-GEMM 3x3 SAME conv, f16 in / f32 accumulate.
// Block tile: M=128 pixels x N=COUT_PAD.  8 waves as 4 M-strips x 2 N-halves;
// each wave: 2 A-fragments (32 pixels) x COUT_PAD/32 B-fragments.
// K = CIN*9 in chunks of 32 staged through LDS with register prefetch.
// EPI: 0 = +bias, store f32+f16   1 = relu(x*g+b), store f16
//      2 = +bias + Threshold2d sigmoid, store f32
// ---------------------------------------------------------------------------
template<int CIN, int COUT_PAD, int COUT, int EPI>
__launch_bounds__(256)
__global__ void conv3x3_wmma(const unsigned short* __restrict__ in16,   // f16 bits
                             const unsigned short* __restrict__ wgt16,  // f16 bits, padded
                             const float* __restrict__ p0,
                             const float* __restrict__ p1,
                             float* __restrict__ out32,
                             unsigned short* __restrict__ out16)
{
    constexpr int KTOT = CIN * 9;
    constexpr int KC   = KTOT / 32;        // 144 / 36 / 36
    constexpr int LDA  = 34;               // padded LDS row stride in halves
    constexpr int NTW  = COUT_PAD / 32;    // B fragments per wave (4 or 2)
    constexpr int NB   = (COUT_PAD * 16) / 256;   // B dwords per thread (8 or 4)

    __shared__ __align__(16) unsigned short As[128 * LDA];
    __shared__ __align__(16) unsigned short Bs[COUT_PAD * LDA];
    unsigned* Bsu = (unsigned*)Bs;

    const int b      = blockIdx.y;
    const int pbase  = blockIdx.x * 128;
    const int tid    = threadIdx.x;
    const int wave   = tid >> 5;
    const int mstrip = wave & 3;           // 4 M-strips of 32 pixels
    const int nhalf  = wave >> 2;          // 2 N-halves of COUT_PAD/2
    const int lane   = tid & 31;
    const int lhi    = lane >> 4;
    const int lm     = lane & 15;

    const unsigned* wgtu = (const unsigned*)wgt16;   // dword view of weights

    v8f acc[2][NTW];
    #pragma unroll
    for (int mi = 0; mi < 2; ++mi)
        #pragma unroll
        for (int i = 0; i < NTW; ++i) { v8f z = {}; acc[mi][i] = z; }

    unsigned short aReg[16];
    unsigned       bReg[NB];

    // ---- stage loader: im2col A (128x32 halves) + weights B (COUT_PADx32)
    auto load_stage = [&](int kc) {
        #pragma unroll
        for (int i = 0; i < 16; ++i) {
            int idx = tid + i * 256;           // over 128*32 elements
            int m = idx >> 5, kk = idx & 31;
            int K  = kc * 32 + kk;
            int ci = K / 9,  r = K % 9;
            int ky = r / 3 - 1, kx = r % 3 - 1;
            int p  = pbase + m;
            int y  = p / WW, x = p % WW;
            int iy = y + ky, ix = x + kx;
            unsigned short v = 0;
            if (p < HW && iy >= 0 && iy < HH && ix >= 0 && ix < WW)
                v = in16[((size_t)(b * CIN + ci)) * HW + iy * WW + ix];
            aReg[i] = v;
        }
        #pragma unroll
        for (int i = 0; i < NB; ++i) {
            int idx = tid + i * 256;           // over COUT_PAD*16 dwords
            int n = idx >> 4, kd = idx & 15;
            bReg[i] = wgtu[(size_t)n * (KTOT / 2) + kc * 16 + kd];
        }
    };
    auto store_stage = [&]() {
        #pragma unroll
        for (int i = 0; i < 16; ++i) {
            int idx = tid + i * 256;
            int m = idx >> 5, kk = idx & 31;
            As[m * LDA + kk] = aReg[i];
        }
        #pragma unroll
        for (int i = 0; i < NB; ++i) {
            int idx = tid + i * 256;
            int n = idx >> 4, kd = idx & 15;
            Bsu[n * (LDA / 2) + kd] = bReg[i];
        }
    };

    load_stage(0);

    for (int kc = 0; kc < KC; ++kc) {
        __syncthreads();                 // LDS free from previous compute
        store_stage();
        __syncthreads();

        if (kc + 1 < KC) load_stage(kc + 1);   // overlap with WMMAs below
        if (kc + 2 < KC) {                     // warm near caches for kc+2
            int n = tid >> 4, kd = tid & 15;
            __builtin_prefetch(&wgtu[(size_t)n * (KTOT / 2) + (kc + 2) * 16 + kd], 0, 3);
        }

        // A fragments: lanes 0-15 hold K{0..7,16..23}, lanes 16-31 K{8..15,24..31}
        union { v16h h; unsigned u[8]; } af[2];
        #pragma unroll
        for (int mi = 0; mi < 2; ++mi) {
            const int am = mstrip * 32 + mi * 16 + lm;
            #pragma unroll
            for (int j = 0; j < 8; ++j) {
                int k0 = ((j < 4) ? 0 : 16) + lhi * 8 + 2 * (j & 3);
                af[mi].u[j] = *(const unsigned*)&As[am * LDA + k0];
            }
        }
        // B fragments (32x16: lanes 0-15 K{0..7,16..23}, 16-31 K{8..15,24..31}), each reused
        // by two consecutive WMMAs.
        #pragma unroll
        for (int nt = 0; nt < NTW; ++nt) {
            union { v16h h; unsigned u[8]; } bf;
            const int bn = nhalf * (COUT_PAD / 2) + nt * 16 + lm;
            #pragma unroll
            for (int j = 0; j < 8; ++j) {
                int k0 = ((j < 4) ? 0 : 16) + lhi * 8 + 2 * (j & 3);
                bf.u[j] = *(const unsigned*)&Bs[bn * LDA + k0];
            }
            #pragma unroll
            for (int mi = 0; mi < 2; ++mi)
                acc[mi][nt] = __builtin_amdgcn_wmma_f32_16x16x32_f16(
                    false, af[mi].h, false, bf.h, (short)0, acc[mi][nt], false, false);
        }
    }

    // ---- epilogue. C/D layout: lane -> N, VGPR r -> M=r (+8 upper half)
    #pragma unroll
    for (int nt = 0; nt < NTW; ++nt) {
        int co = nhalf * (COUT_PAD / 2) + nt * 16 + lm;
        if (co >= COUT) continue;
        #pragma unroll
        for (int mi = 0; mi < 2; ++mi) {
            #pragma unroll
            for (int r = 0; r < 8; ++r) {
                int m = mstrip * 32 + mi * 16 + r + lhi * 8;
                int p = pbase + m;
                if (p >= HW) continue;
                float v = acc[mi][nt][r];
                size_t o = ((size_t)(b * COUT + co)) * HW + p;
                if (EPI == 0) {
                    v += p0[co];
                    out32[o] = v;
                    f16_t h = (f16_t)v;
                    out16[o] = *(unsigned short*)&h;
                } else if (EPI == 1) {
                    v = fmaxf(v * p0[co] + p1[co], 0.f);
                    f16_t h = (f16_t)v;
                    out16[o] = *(unsigned short*)&h;
                } else {
                    v += p0[co];
                    const float lo  = __logf(0.41f)  - 0.1f;
                    const float hi  = __logf(12.29f) - 0.1f;
                    const float inc = (hi - lo) * (1.0f / 19.0f);
                    float thr = lo + (float)(co % 20) * inc;
                    out32[o] = dsigmoid((v - thr) * p1[0]);
                }
            }
        }
    }
}

// ---------------------------------------------------------------------------
// Per-pixel: L2-normalize 60-d vec, cosine vs 10 anchor vectors, sigmoid.
// ---------------------------------------------------------------------------
__global__ void heatmap_kernel(const float* __restrict__ vec,
                               const float* __restrict__ av,
                               const float* __restrict__ lsc,
                               const float* __restrict__ lbi,
                               float* __restrict__ heat)
{
    __shared__ float sav[10 * TEXTC];
    for (int i = threadIdx.x; i < 10 * TEXTC; i += blockDim.x) sav[i] = av[i];
    __syncthreads();
    int p = blockIdx.x * blockDim.x + threadIdx.x;
    int b = blockIdx.y;
    if (p >= HW) return;
    float v[TEXTC];
    float ss = 0.f;
    #pragma unroll
    for (int c = 0; c < TEXTC; ++c) {
        v[c] = vec[((size_t)(b * TEXTC + c)) * HW + p];
        ss += v[c] * v[c];
    }
    float invn = 1.f / (1e-8f + sqrtf(ss));
    float ls = lsc[0], lb = lbi[0];
    #pragma unroll
    for (int k = 0; k < 10; ++k) {
        float s = 0.f;
        #pragma unroll
        for (int c = 0; c < TEXTC; ++c) s += v[c] * sav[k * TEXTC + c];
        heat[((size_t)(b * 10 + k)) * HW + p] = dsigmoid(ls * (s * invn) + lb);
    }
}

// ---------------------------------------------------------------------------
// 3x3 local-max NMS (SAME window).
// ---------------------------------------------------------------------------
__global__ void nms_kernel(const float* __restrict__ heat, float* __restrict__ hnms)
{
    int p  = blockIdx.x * blockDim.x + threadIdx.x;
    int bk = blockIdx.y;                         // b*10 + k
    if (p >= HW) return;
    int y = p / WW, x = p % WW;
    const float* hb = heat + (size_t)bk * HW;
    float h  = hb[p];
    float mx = -1e30f;
    #pragma unroll
    for (int dy = -1; dy <= 1; ++dy)
        #pragma unroll
        for (int dx = -1; dx <= 1; ++dx) {
            int yy = y + dy, xx = x + dx;
            if (yy >= 0 && yy < HH && xx >= 0 && xx < WW)
                mx = fmaxf(mx, hb[yy * WW + xx]);
        }
    hnms[(size_t)bk * HW + p] = (h == mx) ? h : 0.f;
}

// ---------------------------------------------------------------------------
// Per-batch top-200 by iterative block-wide argmax (stable ties -> smallest
// index, matching jax top_k). NMS rewrites its input every launch, so graph
// replays stay deterministic.
// ---------------------------------------------------------------------------
__launch_bounds__(1024)
__global__ void topk_kernel(float* __restrict__ hnms,
                            float* __restrict__ out_scores,
                            float* __restrict__ out_labels,
                            int*   __restrict__ out_idx)
{
    __shared__ float sv[1024];
    __shared__ int   si[1024];
    const int b = blockIdx.x;
    volatile float* v = hnms + (size_t)b * 10 * HW;
    const int N = 10 * HW;
    for (int sel = 0; sel < NPROP; ++sel) {
        float best = -1e30f; int bidx = 0x7fffffff;
        for (int i = threadIdx.x; i < N; i += 1024) {
            float t = v[i];
            if (t > best) { best = t; bidx = i; }
        }
        sv[threadIdx.x] = best; si[threadIdx.x] = bidx;
        __syncthreads();
        for (int s = 512; s > 0; s >>= 1) {
            if (threadIdx.x < (unsigned)s) {
                float ov = sv[threadIdx.x + s]; int oi = si[threadIdx.x + s];
                if (ov > sv[threadIdx.x] ||
                    (ov == sv[threadIdx.x] && oi < si[threadIdx.x])) {
                    sv[threadIdx.x] = ov; si[threadIdx.x] = oi;
                }
            }
            __syncthreads();
        }
        if (threadIdx.x == 0) {
            int w = si[0];
            out_scores[b * NPROP + sel] = sv[0];
            out_labels[b * NPROP + sel] = (float)(w / HW);
            out_idx[b * NPROP + sel]    = w;
            v[w] = -1e30f;
        }
        __threadfence();
        __syncthreads();
    }
}

// ---------------------------------------------------------------------------
// Gather proposal features / positions / class scores + anchor-size encoding.
// ---------------------------------------------------------------------------
__global__ void gather_kernel(const float* __restrict__ lidar,
                              const float* __restrict__ heat,
                              const float* __restrict__ w_enc,
                              const float* __restrict__ bev_pos,
                              const int*   __restrict__ idxbuf,
                              float* __restrict__ qfeat,
                              float* __restrict__ qpos,
                              float* __restrict__ qhm)
{
    int p = blockIdx.x;        // proposal
    int b = blockIdx.y;
    int c = threadIdx.x;       // channel 0..127
    int fl    = idxbuf[b * NPROP + p];
    int label = fl / HW;
    int pos   = fl % HW;
    float la0 = __logf(c_anchors[label][0]);
    float la1 = __logf(c_anchors[label][1]);
    float la2 = __logf(c_anchors[label][2]);
    float qv = lidar[((size_t)(b * HID + c)) * HW + pos]
             + w_enc[c * 3 + 0] * la0 + w_enc[c * 3 + 1] * la1 + w_enc[c * 3 + 2] * la2;
    qfeat[((size_t)(b * HID + c)) * NPROP + p] = qv;
    if (c < 10)
        qhm[((size_t)(b * 10 + c)) * NPROP + p] = heat[((size_t)(b * 10 + c)) * HW + pos];
    if (c < 2)
        qpos[((size_t)(b * NPROP + p)) * 2 + c] = bev_pos[(size_t)pos * 2 + c];
}

// ---------------------------------------------------------------------------
// 10x10 anchor IoU-similarity table.
// ---------------------------------------------------------------------------
__global__ void sims_kernel(float* __restrict__ sims)
{
    int t = threadIdx.x;
    if (t < 100) {
        int i = t / 10, j = t % 10;
        float prod = 1.f;
        #pragma unroll
        for (int d = 0; d < 3; ++d) {
            float a = c_anchors[i][d], c = c_anchors[j][d];
            prod *= fminf(a, c) / fmaxf(a, c);
        }
        sims[t] = prod;
    }
}

// ---------------------------------------------------------------------------
extern "C" void kernel_launch(void* const* d_in, const int* in_sizes, int n_in,
                              void* d_out, int out_size, void* d_ws, size_t ws_size,
                              hipStream_t stream)
{
    const float* feats       = (const float*)d_in[0];   // [2,512,180,180]
    const float* w_shared    = (const float*)d_in[1];   // [128,512,3,3]
    const float* b_shared    = (const float*)d_in[2];   // [128]
    const float* w_bb        = (const float*)d_in[3];   // [128,128,3,3]
    const float* gamma_bb    = (const float*)d_in[4];   // [128]
    const float* beta_bb     = (const float*)d_in[5];   // [128]
    const float* w_hm        = (const float*)d_in[6];   // [60,128,3,3]
    const float* b_hm        = (const float*)d_in[7];   // [60]
    const float* thr_scale   = (const float*)d_in[8];   // [1]
    const float* logit_scale = (const float*)d_in[9];   // [1]
    const float* logit_bias  = (const float*)d_in[10];  // [1]
    const float* w_enc       = (const float*)d_in[11];  // [128,3]
    const float* anchor_vecs = (const float*)d_in[12];  // [10,60]
    const float* bev_pos     = (const float*)d_in[13];  // [HW,2]

    // workspace layout -------------------------------------------------------
    // f32: lidar_feat | vec | nms | idx     f16: feats16 | w1 | w2 | w3 | lidar16 | x16
    float* ws        = (float*)d_ws;
    float* ws_lidar  = ws;                                     // 2*128*HW f32
    float* ws_vec    = ws_lidar + (size_t)HB * HID * HW;       // 2*60*HW  f32
    float* ws_nms    = ws_vec   + (size_t)HB * TEXTC * HW;     // 2*10*HW  f32
    int*   ws_idx    = (int*)(ws_nms + (size_t)HB * 10 * HW);  // 2*200    i32
    f16_t* ws_h      = (f16_t*)(ws_idx + HB * NPROP);
    f16_t* ws_feat16 = ws_h;                                   // 2*512*HW
    f16_t* ws_w1     = ws_feat16 + (size_t)HB * CIN1 * HW;     // 128*4608
    f16_t* ws_w2     = ws_w1 + (size_t)128 * CIN1 * 9;         // 128*1152
    f16_t* ws_w3     = ws_w2 + (size_t)128 * HID * 9;          // 64*1152 (padded)
    f16_t* ws_lid16  = ws_w3 + (size_t)64 * HID * 9;           // 2*128*HW
    f16_t* ws_x16    = ws_lid16 + (size_t)HB * HID * HW;       // 2*128*HW

    // output layout (flat, reference return order) --------------------------
    float* out      = (float*)d_out;
    float* o_heat   = out;                                   // 2*10*HW
    float* o_scores = o_heat   + (size_t)HB * 10 * HW;       // 400
    float* o_labels = o_scores + HB * NPROP;                 // 400
    float* o_qfeat  = o_labels + HB * NPROP;                 // 2*128*200
    float* o_qpos   = o_qfeat  + (size_t)HB * HID * NPROP;   // 800
    float* o_qhm    = o_qpos   + (size_t)HB * NPROP * 2;     // 4000
    float* o_sims   = o_qhm    + (size_t)HB * 10 * NPROP;    // 100

    // ---- one-time f32 -> f16 converts (feats + padded weights)
    long nfeat = (long)HB * CIN1 * HW;
    cvt_f16_kernel<<<8192, 256, 0, stream>>>(feats, ws_feat16, nfeat);
    cvt_wgt_kernel<<<1024, 256, 0, stream>>>(w_shared, ws_w1, 128, 128, CIN1 * 9);
    cvt_wgt_kernel<<<256, 256, 0, stream>>>(w_bb, ws_w2, 128, 128, HID * 9);
    cvt_wgt_kernel<<<256, 256, 0, stream>>>(w_hm, ws_w3, TEXTC, 64, HID * 9);

    dim3 gconv((HW + 127) / 128, HB);

    // conv1: 512 -> 128, +bias, store f32 (for gather) + f16 (for conv2)
    conv3x3_wmma<CIN1, 128, 128, 0><<<gconv, 256, 0, stream>>>(
        (const unsigned short*)ws_feat16, (const unsigned short*)ws_w1,
        b_shared, b_shared, ws_lidar, (unsigned short*)ws_lid16);

    // conv2 (BasicBlock2D): 128 -> 128, relu(x*gamma+beta), store f16
    conv3x3_wmma<HID, 128, 128, 1><<<gconv, 256, 0, stream>>>(
        (const unsigned short*)ws_lid16, (const unsigned short*)ws_w2,
        gamma_bb, beta_bb, nullptr, (unsigned short*)ws_x16);

    // conv3: 128 -> 60 (N pad 64), +bias + Threshold2d sigmoid, store f32
    conv3x3_wmma<HID, 64, TEXTC, 2><<<gconv, 256, 0, stream>>>(
        (const unsigned short*)ws_x16, (const unsigned short*)ws_w3,
        b_hm, thr_scale, ws_vec, nullptr);

    // normalize + anchor cosine + sigmoid -> heatmap
    dim3 ghm((HW + 255) / 256, HB);
    heatmap_kernel<<<ghm, 256, 0, stream>>>(ws_vec, anchor_vecs,
                                            logit_scale, logit_bias, o_heat);

    // 3x3 NMS
    dim3 gnms((HW + 255) / 256, HB * 10);
    nms_kernel<<<gnms, 256, 0, stream>>>(o_heat, ws_nms);

    // per-batch top-200
    topk_kernel<<<HB, 1024, 0, stream>>>(ws_nms, o_scores, o_labels, ws_idx);

    // proposal gather + anchor-size encoding
    dim3 gg(NPROP, HB);
    gather_kernel<<<gg, 128, 0, stream>>>(ws_lidar, o_heat, w_enc, bev_pos,
                                          ws_idx, o_qfeat, o_qpos, o_qhm);

    // anchor IoU-similarity table
    sims_kernel<<<1, 128, 0, stream>>>(o_sims);
}
